// Mamba2PathReader_46042049413262
// MI455X (gfx1250) — matrix operand order repaired and checked by reading, not compiled, not run on hardware
//
#include <hip/hip_runtime.h>
#include <math.h>

// ---------------- model dims ----------------
#define Lc    4
#define Hc    1024
#define Ic    2048
#define Nc    128
#define NHc   8
#define HDc   256
#define Kc    4
#define PROJc (2*Ic + 2*Nc + NHc)   // 4360
#define CONVD (Ic + 2*Nc)           // 2304
#define Bc    4
#define Tc    2048
#define MT    (Bc*Tc)               // 8192
#define EPSf  1e-5f

typedef __bf16 bf16_t;
typedef bf16_t bf16x8  __attribute__((ext_vector_type(8)));
typedef bf16_t bf16x16 __attribute__((ext_vector_type(16)));
typedef float  f32x8   __attribute__((ext_vector_type(8)));

static __device__ __forceinline__ bf16_t f2bf(float f) {
  unsigned u = __builtin_bit_cast(unsigned, f);
  unsigned r = (u + 0x7FFFu + ((u >> 16) & 1u)) >> 16;
  unsigned short s = (unsigned short)r;
  return __builtin_bit_cast(bf16_t, s);
}
static __device__ __forceinline__ float siluf(float x) {
  return x / (1.0f + expf(-x));
}

// ---------------- fp32 -> bf16 cast ----------------
__global__ void __launch_bounds__(256)
cast_bf16_kernel(const float* __restrict__ src, bf16_t* __restrict__ dst, int n) {
  int i = blockIdx.x * 256 + threadIdx.x;
  if (i < n) dst[i] = f2bf(src[i]);
}

// ---------------- RMSNorm (row length = dim), bf16 out ----------------
__global__ void __launch_bounds__(256)
rmsnorm_bf16_kernel(const float* __restrict__ x, const float* __restrict__ w,
                    bf16_t* __restrict__ out, int dim) {
  __shared__ float red[256];
  const int row = blockIdx.x;
  const float* xr = x + (size_t)row * dim;
  float p = 0.f;
  for (int i = threadIdx.x; i < dim; i += 256) { float v = xr[i]; p += v * v; }
  red[threadIdx.x] = p; __syncthreads();
  for (int s = 128; s > 0; s >>= 1) {
    if (threadIdx.x < s) red[threadIdx.x] += red[threadIdx.x + s];
    __syncthreads();
  }
  float scale = rsqrtf(red[0] / (float)dim + EPSf);
  bf16_t* orow = out + (size_t)row * dim;
  for (int i = threadIdx.x; i < dim; i += 256)
    orow[i] = f2bf(w[i] * xr[i] * scale);
}

// ---------------- RMSNorm, f32 out (final norm) ----------------
__global__ void __launch_bounds__(256)
rmsnorm_f32_kernel(const float* __restrict__ x, const float* __restrict__ w,
                   float* __restrict__ out, int dim) {
  __shared__ float red[256];
  const int row = blockIdx.x;
  const float* xr = x + (size_t)row * dim;
  float p = 0.f;
  for (int i = threadIdx.x; i < dim; i += 256) { float v = xr[i]; p += v * v; }
  red[threadIdx.x] = p; __syncthreads();
  for (int s = 128; s > 0; s >>= 1) {
    if (threadIdx.x < s) red[threadIdx.x] += red[threadIdx.x + s];
    __syncthreads();
  }
  float scale = rsqrtf(red[0] / (float)dim + EPSf);
  float* orow = out + (size_t)row * dim;
  for (int i = threadIdx.x; i < dim; i += 256)
    orow[i] = w[i] * xr[i] * scale;
}

// ---------------- WMMA bf16 GEMM: C[M,Nreal] = A[M,K] * W[Nreal,K]^T ----------------
// Block = 256 threads (8 waves, 2x4 grid). Block tile 64(M) x 256(N).
// Each wave: 32x64 C tile = 2 A-frags x 4 B-frags = 8 WMMAs per 32-K step.
__global__ void __launch_bounds__(256)
wmma_gemm_bf16f32(const bf16_t* __restrict__ A, const bf16_t* __restrict__ W,
                  float* __restrict__ C, int M, int Nreal, int K) {
  __shared__ bf16_t As[64][32];    // M x K tile (rows 64B)
  __shared__ bf16_t Bs[256][48];   // N x K tile, padded rows (96B) for alignment/banks

  const int tid  = threadIdx.x;
  const int wv   = tid >> 5;
  const int lane = tid & 31;
  const int hl   = lane >> 4;      // half-wave
  const int lr   = lane & 15;
  const int wrow = wv >> 2;        // 0..1  -> M offset wrow*32
  const int wcol = wv & 3;         // 0..3  -> N offset wcol*64
  const int mbase = blockIdx.x * 64;
  const int nbase = blockIdx.y * 256;

  // cooperative-load mappings
  const int arow = tid >> 2;            // 0..63
  const int acol = (tid & 3) * 8;       // 0,8,16,24
  const int wn   = tid;                 // 0..255 : one B row per thread

  f32x8 acc[2][4];
#pragma unroll
  for (int ms = 0; ms < 2; ++ms)
#pragma unroll
    for (int ns = 0; ns < 4; ++ns) acc[ms][ns] = (f32x8){};

  const int ng = nbase + wn;
  const bf16_t* wrowp = W + (size_t)ng * K;
  const bf16_t* arowp = A + (size_t)(mbase + arow) * K;

  for (int k0 = 0; k0 < K; k0 += 32) {
    // --- stage A tile (64x32): 8 contiguous bf16 per thread ---
    bf16x8 av = *(const bf16x8*)(arowp + k0 + acol);
    *(bf16x8*)&As[arow][acol] = av;
    if (k0 + 32 < K)
      __builtin_prefetch(arowp + k0 + 32 + acol, 0, 1);

    // --- stage W tile as Bs[n][k] (256 x 32): one 32-elem row per thread ---
    bf16x8 w0 = {}, w1 = {}, w2 = {}, w3 = {};
    if (ng < Nreal) {
      const bf16x8* src = (const bf16x8*)(wrowp + k0);
      w0 = src[0]; w1 = src[1]; w2 = src[2]; w3 = src[3];
      if (k0 + 32 < K)
        __builtin_prefetch(wrowp + k0 + 32, 0, 1);
    }
    *(bf16x8*)&Bs[wn][0]  = w0;
    *(bf16x8*)&Bs[wn][8]  = w1;
    *(bf16x8*)&Bs[wn][16] = w2;
    *(bf16x8*)&Bs[wn][24] = w3;
    __syncthreads();

    // --- gather 2 A fragments (documented 16-bit 16x32 layout) ---
    bf16x16 afrag[2];
#pragma unroll
    for (int ms = 0; ms < 2; ++ms) {
      const int r = wrow * 32 + ms * 16 + lr;
      bf16x8 alo = *(const bf16x8*)&As[r][hl * 8];
      bf16x8 ahi = *(const bf16x8*)&As[r][16 + hl * 8];
#pragma unroll
      for (int i = 0; i < 8; ++i) { afrag[ms][i] = alo[i]; afrag[ms][i + 8] = ahi[i]; }
    }
    // --- gather 4 B fragments: column n, K = hl*8 .. +7 and 16+hl*8 .. +7 ---
    bf16x16 bfrag[4];
#pragma unroll
    for (int ns = 0; ns < 4; ++ns) {
      const int bn = wcol * 64 + ns * 16 + lr;
      bf16x8 blo = *(const bf16x8*)&Bs[bn][hl * 8];
      bf16x8 bhi = *(const bf16x8*)&Bs[bn][16 + hl * 8];
#pragma unroll
      for (int i = 0; i < 8; ++i) { bfrag[ns][i] = blo[i]; bfrag[ns][i + 8] = bhi[i]; }
    }

    // --- 8 WMMAs: register-blocked outer product of fragments ---
#pragma unroll
    for (int ms = 0; ms < 2; ++ms)
#pragma unroll
      for (int ns = 0; ns < 4; ++ns)
        acc[ms][ns] = __builtin_amdgcn_wmma_f32_16x16x32_bf16(
            false, afrag[ms], false, bfrag[ns], (short)0, acc[ms][ns], false, false);

    __syncthreads();
  }

  // --- store C ---
#pragma unroll
  for (int ms = 0; ms < 2; ++ms) {
#pragma unroll
    for (int ns = 0; ns < 4; ++ns) {
      const int cn = nbase + wcol * 64 + ns * 16 + lr;
      if (cn < Nreal) {
#pragma unroll
        for (int j = 0; j < 8; ++j) {
          const int cm = mbase + wrow * 32 + ms * 16 + hl * 8 + j;
          C[(size_t)cm * Nreal + cn] = acc[ms][ns][j];
        }
      }
    }
  }
}

// ---------------- causal depthwise conv1d (K=4) + bias + SiLU ----------------
__global__ void __launch_bounds__(256)
conv_silu_kernel(const float* __restrict__ proj, const float* __restrict__ cw,
                 const float* __restrict__ cb, float* __restrict__ out) {
  long idx = (long)blockIdx.x * 256 + threadIdx.x;
  const long total = (long)MT * CONVD;
  if (idx >= total) return;
  int  c  = (int)(idx % CONVD);
  long bt = idx / CONVD;
  int  t  = (int)(bt % Tc);
  long b0 = bt - t;                 // b*T
  float acc = cb[c];
#pragma unroll
  for (int j = 0; j < Kc; ++j) {
    int tt = t - (Kc - 1) + j;
    if (tt >= 0)
      acc += cw[c * Kc + j] * proj[(b0 + tt) * (long)PROJc + Ic + c];
  }
  out[idx] = siluf(acc);
}

// ---------------- dt = softplus(dt_raw + bias); dA = exp(dt * -exp(A_log)) ----------------
__global__ void __launch_bounds__(256)
dt_dA_kernel(const float* __restrict__ proj, const float* __restrict__ dt_bias,
             const float* __restrict__ A_log, float* __restrict__ dtv,
             float* __restrict__ dA) {
  int idx = blockIdx.x * 256 + threadIdx.x;
  if (idx >= MT * NHc) return;
  int nh = idx % NHc;
  long bt = idx / NHc;
  float d = proj[bt * (long)PROJc + (Ic + CONVD) + nh] + dt_bias[nh];
  float sp = (d > 20.f) ? d : log1pf(expf(d));
  dtv[idx] = sp;
  dA[idx]  = expf(-expf(A_log[nh]) * sp);
}

// ---------------- SSM scan: one block per (b, nh); thread d keeps S[d][0..127] in VGPRs ----------------
__global__ void __launch_bounds__(256)
scan_kernel(const float* __restrict__ convout, const float* __restrict__ dA,
            const float* __restrict__ dtv, const float* __restrict__ Dp,
            float* __restrict__ y) {
  const int b  = blockIdx.x / NHc;
  const int nh = blockIdx.x % NHc;
  const int d  = threadIdx.x;            // 0..255 (HD)
  __shared__ float Bsm[Nc];
  __shared__ float Csm[Nc];
  float S[Nc];
#pragma unroll
  for (int n = 0; n < Nc; ++n) S[n] = 0.f;
  const float dcoef = Dp[nh];

  for (int t = 0; t < Tc; ++t) {
    const long bt = (long)b * Tc + t;
    const float* row = convout + bt * (long)CONVD;
    if (threadIdx.x < Nc)       Bsm[threadIdx.x]       = row[Ic + threadIdx.x];
    else                        Csm[threadIdx.x - Nc]  = row[Ic + Nc + (threadIdx.x - Nc)];
    float xv = row[nh * HDc + d];
    float da = dA[bt * NHc + nh];
    float xd = xv * dtv[bt * NHc + nh];
    __syncthreads();
    float accv = 0.f;
#pragma unroll
    for (int n = 0; n < Nc; ++n) {
      S[n] = da * S[n] + xd * Bsm[n];
      accv += S[n] * Csm[n];
    }
    y[bt * (long)Ic + nh * HDc + d] = accv + dcoef * xv;
    __syncthreads();
  }
}

// ---------------- gated RMSNorm: out_bf = gw * (y*silu(z)) * rsqrt(mean((y*silu(z))^2)+eps) ----------------
__global__ void __launch_bounds__(256)
gated_norm_kernel(const float* __restrict__ y, const float* __restrict__ proj,
                  const float* __restrict__ gw, bf16_t* __restrict__ out) {
  __shared__ float red[256];
  const int row = blockIdx.x;
  const float* yr = y + (size_t)row * Ic;
  const float* zr = proj + (size_t)row * PROJc;   // z = first Ic columns
  float p = 0.f;
  for (int i = threadIdx.x; i < Ic; i += 256) {
    float u = yr[i] * siluf(zr[i]);
    p += u * u;
  }
  red[threadIdx.x] = p; __syncthreads();
  for (int s = 128; s > 0; s >>= 1) {
    if (threadIdx.x < s) red[threadIdx.x] += red[threadIdx.x + s];
    __syncthreads();
  }
  float scale = rsqrtf(red[0] / (float)Ic + EPSf);
  bf16_t* orow = out + (size_t)row * Ic;
  for (int i = threadIdx.x; i < Ic; i += 256) {
    float u = yr[i] * siluf(zr[i]);
    orow[i] = f2bf(gw[i] * u * scale);
  }
}

// ---------------- residual add ----------------
__global__ void __launch_bounds__(256)
add_kernel(float* __restrict__ h, const float* __restrict__ m, int n) {
  int i = blockIdx.x * 256 + threadIdx.x;
  if (i < n) h[i] += m[i];
}

// ---------------- mean pool over T ----------------
__global__ void __launch_bounds__(256)
pool_kernel(const float* __restrict__ hn, float* __restrict__ pooled) {
  int idx = blockIdx.x * 256 + threadIdx.x;
  if (idx >= Bc * Hc) return;
  int b = idx / Hc, c = idx % Hc;
  float s = 0.f;
  for (int t = 0; t < Tc; ++t)
    s += hn[((long)b * Tc + t) * Hc + c];
  pooled[idx] = s * (1.0f / Tc);
}

// ---------------- head: out = pooled @ head_w^T + head_b ----------------
__global__ void __launch_bounds__(256)
head_kernel(const float* __restrict__ pooled, const float* __restrict__ hw,
            const float* __restrict__ hb, float* __restrict__ out) {
  int idx = blockIdx.x * 256 + threadIdx.x;
  if (idx >= Bc * Hc) return;
  int b = idx / Hc, j = idx % Hc;
  float acc = hb[j];
  const float* pr = pooled + (size_t)b * Hc;
  const float* wr = hw + (size_t)j * Hc;
  for (int k = 0; k < Hc; ++k) acc += pr[k] * wr[k];
  out[idx] = acc;
}

// ============================================================================
extern "C" void kernel_launch(void* const* d_in, const int* in_sizes, int n_in,
                              void* d_out, int out_size, void* d_ws, size_t ws_size,
                              hipStream_t stream) {
  const float* x       = (const float*)d_in[0];
  const float* in_w    = (const float*)d_in[1];
  const float* conv_w  = (const float*)d_in[2];
  const float* conv_b  = (const float*)d_in[3];
  const float* dt_bias = (const float*)d_in[4];
  const float* A_log   = (const float*)d_in[5];
  const float* Dp      = (const float*)d_in[6];
  const float* gnorm_w = (const float*)d_in[7];
  const float* out_w   = (const float*)d_in[8];
  const float* ln_w    = (const float*)d_in[9];
  const float* normf_w = (const float*)d_in[10];
  const float* head_w  = (const float*)d_in[11];
  const float* head_b  = (const float*)d_in[12];

  // ---- workspace carve-up ----
  char* p = (char*)d_ws;
  auto carve = [&](size_t bytes) {
    char* r = p;
    p += (bytes + 255) & ~(size_t)255;
    return (void*)r;
  };
  float*  h       = (float*) carve((size_t)MT * Hc * 4);
  bf16_t* xn      = (bf16_t*)carve((size_t)MT * Hc * 2);
  float*  proj    = (float*) carve((size_t)MT * PROJc * 4);
  float*  convout = (float*) carve((size_t)MT * CONVD * 4);
  float*  dtv     = (float*) carve((size_t)MT * NHc * 4);
  float*  dAb     = (float*) carve((size_t)MT * NHc * 4);
  float*  ybuf    = (float*) carve((size_t)MT * Ic * 4);
  bf16_t* ybf     = (bf16_t*)carve((size_t)MT * Ic * 2);
  float*  mix     = (float*) carve((size_t)MT * Hc * 4);   // reused as hn after layers
  float*  pooled  = (float*) carve((size_t)Bc * Hc * 4);
  bf16_t* wbf_in  = (bf16_t*)carve((size_t)PROJc * Hc * 2);
  bf16_t* wbf_out = (bf16_t*)carve((size_t)Hc * Ic * 2);

  // h = x (residual stream)
  hipMemcpyAsync(h, x, (size_t)MT * Hc * 4, hipMemcpyDeviceToDevice, stream);

  const int inw_elems  = PROJc * Hc;   // per-layer in_proj weight
  const int outw_elems = Hc * Ic;      // per-layer out_proj weight

  for (int l = 0; l < Lc; ++l) {
    // cast this layer's weights to bf16
    cast_bf16_kernel<<<(inw_elems + 255) / 256, 256, 0, stream>>>(
        in_w + (size_t)l * inw_elems, wbf_in, inw_elems);
    cast_bf16_kernel<<<(outw_elems + 255) / 256, 256, 0, stream>>>(
        out_w + (size_t)l * outw_elems, wbf_out, outw_elems);

    // xn = rmsnorm(h, ln_w[l]) in bf16
    rmsnorm_bf16_kernel<<<MT, 256, 0, stream>>>(h, ln_w + (size_t)l * Hc, xn, Hc);

    // proj = xn @ in_w[l]^T   (M=8192, K=1024, N=4360)
    {
      dim3 grid(MT / 64, (PROJc + 255) / 256);
      wmma_gemm_bf16f32<<<grid, 256, 0, stream>>>(xn, wbf_in, proj, MT, PROJc, Hc);
    }

    // conv + SiLU on xBC columns
    {
      long total = (long)MT * CONVD;
      conv_silu_kernel<<<(int)((total + 255) / 256), 256, 0, stream>>>(
          proj, conv_w + (size_t)l * CONVD * Kc, conv_b + (size_t)l * CONVD, convout);
    }

    // dt / dA
    dt_dA_kernel<<<(MT * NHc + 255) / 256, 256, 0, stream>>>(
        proj, dt_bias + (size_t)l * NHc, A_log + (size_t)l * NHc, dtv, dAb);

    // sequential SSM scan
    scan_kernel<<<Bc * NHc, HDc, 0, stream>>>(convout, dAb, dtv,
                                              Dp + (size_t)l * NHc, ybuf);

    // gated RMSNorm -> bf16
    gated_norm_kernel<<<MT, 256, 0, stream>>>(ybuf, proj,
                                              gnorm_w + (size_t)l * Ic, ybf);

    // mix = ybf @ out_w[l]^T   (M=8192, K=2048, N=1024)
    {
      dim3 grid(MT / 64, Hc / 256);
      wmma_gemm_bf16f32<<<grid, 256, 0, stream>>>(ybf, wbf_out, mix, MT, Hc, Ic);
    }

    // h += mix
    add_kernel<<<(MT * Hc + 255) / 256, 256, 0, stream>>>(h, mix, MT * Hc);
  }

  // final norm (reuse mix buffer as hn), pool, head
  rmsnorm_f32_kernel<<<MT, 256, 0, stream>>>(h, normf_w, mix, Hc);
  pool_kernel<<<(Bc * Hc + 255) / 256, 256, 0, stream>>>(mix, pooled);
  head_kernel<<<(Bc * Hc + 255) / 256, 256, 0, stream>>>(pooled, head_w, head_b,
                                                         (float*)d_out);
}
